// AttentionRNNCell_78675210928359
// MI455X (gfx1250) — matrix operand order repaired and checked by reading, not compiled, not run on hardware
//
#include <hip/hip_runtime.h>
#include <math.h>

// ---------------------------------------------------------------------------
// AttentionRNNCell for MI455X (gfx1250, wave32, WMMA). Sizes fixed by ref.
// ---------------------------------------------------------------------------
constexpr int NB = 128;   // batch
constexpr int NT = 1024;  // time
constexpr int NA = 512;   // annot dim
constexpr int NR = 1024;  // rnn dim
constexpr int NM = 256;   // memory dim
constexpr int ND = 256;   // attn dim
constexpr int NF = 32;    // loc filters
constexpr int NK = 31;    // conv kernel

// Packed operand layout ("fragment-slab"): identical to the LDS image the WMMA
// fragment loads consume, so staging is a contiguous b128 memcpy.
//   B (KxN): slab (nb,s) holds n in [nb*256, nb*256+256), k in [s*32, s*32+32)
//            at out[slab*10240 + nn*40 + kk]   (10240 ushorts per slab)
//   A (MxK): slab (mb,s) holds m in [mb*32 ...), k in [s*32 ...)
//            at out[slab*1280 + mm*40 + kk]    (1280 ushorts per slab)
// kk in [32,40) is padding (never read by fragments; pad keeps LDS banks clean).

typedef __attribute__((ext_vector_type(16))) __bf16 v16bf;
typedef __attribute__((ext_vector_type(8)))  float  v8f;

union BF16Frag {
  v16bf v;
  unsigned short u16[16];
  unsigned int   u32[8];
};

__device__ __forceinline__ unsigned short f2bf(float f) {
  unsigned int u = __float_as_uint(f);
  unsigned int r = u + 0x7fffu + ((u >> 16) & 1u);   // round-to-nearest-even
  return (unsigned short)(r >> 16);
}
__device__ __forceinline__ unsigned int pk2(float a, float b) {
  return (unsigned int)f2bf(a) | ((unsigned int)f2bf(b) << 16);
}
// Packed-A element index (for scalar producers of activations).
__device__ __forceinline__ size_t apack_idx(int m, int k, int KS) {
  return ((size_t)((m >> 5) * KS + (k >> 5)) * 32 + (m & 31)) * 40 + (k & 31);
}

// ---------------------------------------------------------------------------
// Pack weight W (N x K fp32, row-major) into B fragment-slab bf16 layout
// (B = W^T, K x N logical). grid (K/32, N/32), 256 threads.
// ---------------------------------------------------------------------------
__global__ __launch_bounds__(256) void k_packB(
    const float* __restrict__ in, unsigned short* __restrict__ out, int N, int K) {
  __shared__ float tile[32][33];
  int k0 = blockIdx.x * 32, n0 = blockIdx.y * 32;
  int j = threadIdx.x & 31, i0 = threadIdx.x >> 5;
  for (int i = i0; i < 32; i += 8)
    tile[i][j] = in[(size_t)(n0 + i) * K + k0 + j];   // coalesced in k
  __syncthreads();
  const int KS = K >> 5;
  const size_t slab = (size_t)((n0 >> 8) * KS + (k0 >> 5)) * 10240;
  int i = threadIdx.x >> 3, q = threadIdx.x & 7;      // row i, 4 k-values each
  uint2 w;
  w.x = pk2(tile[i][q * 4 + 0], tile[i][q * 4 + 1]);
  w.y = pk2(tile[i][q * 4 + 2], tile[i][q * 4 + 3]);
  *(uint2*)&out[slab + (size_t)((n0 & 255) + i) * 40 + q * 4] = w;
}

// ---------------------------------------------------------------------------
// Build packed-A activations: x = concat(memory, context) (KS=24) and h0 (KS=32).
// ---------------------------------------------------------------------------
__global__ __launch_bounds__(256) void k_prep(
    const float* __restrict__ mem, const float* __restrict__ ctx,
    const float* __restrict__ h0,
    unsigned short* __restrict__ x_pk, unsigned short* __restrict__ h0_pk) {
  int idx = blockIdx.x * 256 + threadIdx.x;
  constexpr int XW = NM + NA;  // 768
  if (idx < NB * XW) {
    int b = idx / XW, c = idx % XW;
    float v = (c < NM) ? mem[b * NM + c] : ctx[b * NA + (c - NM)];
    x_pk[apack_idx(b, c, XW >> 5)] = f2bf(v);
  }
  if (idx < NB * NR) {
    int b = idx >> 10, j = idx & (NR - 1);
    h0_pk[apack_idx(b, j, NR >> 5)] = f2bf(h0[idx]);
  }
}

// ---------------------------------------------------------------------------
// bf16 WMMA GEMM on packed operands: C(MxN) f32 = A*B + bias[n].
// Block tile 32(M) x 256(N), 8 waves (2x4), K-step 32. grid (M/32, N/256).
// Staging per step: A = 2.5KB contiguous copy, B = 20KB contiguous copy.
// ---------------------------------------------------------------------------
__global__ __launch_bounds__(256) void k_gemm_bf16(
    const unsigned short* __restrict__ Apk, const unsigned short* __restrict__ Bpk,
    const float* __restrict__ bias, float* __restrict__ Cg,
    int M, int N, int K) {
  __shared__ __align__(16) unsigned short sA[32 * 40];
  __shared__ __align__(16) unsigned short sB[256 * 40];
  const int tid = threadIdx.x, lane = tid & 31, wv = tid >> 5;
  const int bx = blockIdx.x, by = blockIdx.y;
  const int wmRow = (wv >> 2) * 16;  // 0 or 16
  const int wnCol = (wv & 3) * 64;   // 0,64,128,192
  const int KS = K >> 5;

  const v8f vz = {0.f, 0.f, 0.f, 0.f, 0.f, 0.f, 0.f, 0.f};
  v8f acc[4];
#pragma unroll
  for (int nt = 0; nt < 4; ++nt) acc[nt] = vz;

  for (int s = 0; s < KS; ++s) {
    __syncthreads();
    {  // contiguous b128 staging (packed global image == LDS image)
      const uint4* srcA = (const uint4*)(Apk + (size_t)(bx * KS + s) * 1280);
      const uint4* srcB = (const uint4*)(Bpk + (size_t)(by * KS + s) * 10240);
      uint4* dA = (uint4*)sA;
      uint4* dB = (uint4*)sB;
      if (tid < 160) dA[tid] = srcA[tid];                       // 2560 B
#pragma unroll
      for (int i = 0; i < 5; ++i) dB[tid + i * 256] = srcB[tid + i * 256];  // 20480 B
      if (s + 1 < KS)
        __builtin_prefetch(Bpk + (size_t)(by * KS + s + 1) * 10240, 0, 1);
    }
    __syncthreads();

    // A fragment: 16x32 bf16 (ISA 7.12.2; half-lanes offset K by 8).
    BF16Frag fa;
    {
      const unsigned short* row = sA + (wmRow + (lane & 15)) * 40 + (lane >> 4) * 8;
#pragma unroll
      for (int g = 0; g < 2; ++g)
#pragma unroll
        for (int p = 0; p < 4; ++p)
          fa.u32[g * 4 + p] = *(const unsigned int*)(row + g * 16 + p * 2);
    }
#pragma unroll
    for (int nt = 0; nt < 4; ++nt) {
      // B fragment: 32x16 bf16 (half-lanes offset K by 8, same k order as A);
      // [n][k] layout -> two 8-value K runs per lane.
      BF16Frag fb;
      const unsigned short* row =
          sB + (wnCol + nt * 16 + (lane & 15)) * 40 + (lane >> 4) * 8;
#pragma unroll
      for (int p = 0; p < 8; ++p)
        fb.u32[p] = *(const unsigned int*)(row + (p >> 2) * 16 + (p & 3) * 2);
      acc[nt] = __builtin_amdgcn_wmma_f32_16x16x32_bf16(
          false, fa.v, false, fb.v, (short)0, acc[nt], false, false);
    }
  }

  // C/D layout: lane n = lane&15, VGPR r -> m = r + 8*(lane>>4).
#pragma unroll
  for (int nt = 0; nt < 4; ++nt) {
    int n = by * 256 + wnCol + nt * 16 + (lane & 15);
    float bs = bias ? bias[n] : 0.f;
#pragma unroll
    for (int r = 0; r < 8; ++r) {
      int m = bx * 32 + wmRow + r + 8 * (lane >> 4);
      Cg[(size_t)m * N + n] = acc[nt][r] + bs;
    }
  }
}

// ---------------------------------------------------------------------------
// GRU elementwise (gate order r,z,n; biases folded into gi/gh by the GEMMs).
// Writes h fp32 (output 0) and packed-A bf16 h (KS=32) for the pq GEMM.
// ---------------------------------------------------------------------------
__global__ __launch_bounds__(256) void k_gru(
    const float* __restrict__ gi, const float* __restrict__ gh,
    const float* __restrict__ h0, float* __restrict__ h_out,
    unsigned short* __restrict__ h_pk) {
  int idx = blockIdx.x * 256 + threadIdx.x;
  if (idx >= NB * NR) return;
  int b = idx >> 10, j = idx & (NR - 1);
  const float* gib = gi + (size_t)b * 3 * NR;
  const float* ghb = gh + (size_t)b * 3 * NR;
  float r = 1.f / (1.f + __expf(-(gib[j] + ghb[j])));
  float z = 1.f / (1.f + __expf(-(gib[NR + j] + ghb[NR + j])));
  float n = tanhf(gib[2 * NR + j] + r * ghb[2 * NR + j]);
  float h = (1.f - z) * n + z * h0[idx];
  h_out[idx] = h;
  h_pk[apack_idx(b, j, NR >> 5)] = f2bf(h);
}

// ---------------------------------------------------------------------------
// Location conv: atten (B,2,T) * conv_w (F,2,31), pad 15 -> loc (B,T,F) bf16.
// grid (B, T/128), 256 threads.
// ---------------------------------------------------------------------------
__global__ __launch_bounds__(256) void k_locconv(
    const float* __restrict__ atten, const float* __restrict__ cw,
    unsigned short* __restrict__ loc_bf) {
  __shared__ float sAtt[2][128 + 30];
  __shared__ float sW[NF * 2 * NK];
  int b = blockIdx.x, t0 = blockIdx.y * 128, tid = threadIdx.x;
  for (int i = tid; i < NF * 2 * NK; i += 256) sW[i] = cw[i];
  for (int i = tid; i < 2 * 158; i += 256) {
    int c = i / 158, tt = i % 158;
    int t = t0 + tt - 15;
    sAtt[c][tt] = (t >= 0 && t < NT) ? atten[((size_t)b * 2 + c) * NT + t] : 0.f;
  }
  __syncthreads();
  for (int i = tid; i < 128 * NF; i += 256) {
    int f = i & (NF - 1), tt = i >> 5;
    const float* w0 = sW + f * 2 * NK;
    float s = 0.f;
#pragma unroll
    for (int k = 0; k < NK; ++k) {
      s = fmaf(sAtt[0][tt + k], w0[k], s);
      s = fmaf(sAtt[1][tt + k], w0[NK + k], s);
    }
    loc_bf[((size_t)b * NT + t0 + tt) * NF + f] = f2bf(s);
  }
}

// ---------------------------------------------------------------------------
// Fused align kernel:
//   raw[b,t] = sum_d v[d]*tanh( pq[b,d] + (annots@aW^T)[t,d] + (loc@locW^T)[t,d]
//                               + a_b[d] + loc_b[d] )
// One block = (batch b, 32 t-rows). annots -> bf16 LDS once (vectorized),
// 16 WMMA K-steps over A=512 + 1 K-step for the F=32 loc projection.
// grid (B, T/32), 256 threads (8 waves: 2(M) x 4(N) over D=256).
// ---------------------------------------------------------------------------
__global__ __launch_bounds__(256) void k_align(
    const float* __restrict__ annots,
    const unsigned short* __restrict__ aWp,    // packed B slabs, KS=16
    const unsigned short* __restrict__ locWp,  // packed B slab, KS=1
    const unsigned short* __restrict__ loc_bf, // (B,T,NF) bf16
    const float* __restrict__ pq,              // (B,ND), includes q_b
    const float* __restrict__ a_b, const float* __restrict__ loc_b,
    const float* __restrict__ v_w,
    float* __restrict__ align_raw) {
  __shared__ __align__(16) unsigned short sA[32 * 520];   // annots tile bf16
  __shared__ __align__(16) unsigned short sLoc[32 * 40];  // loc tile bf16
  __shared__ __align__(16) unsigned short sB[256 * 40];   // B slab [n][k]
  __shared__ float sPB[ND];
  __shared__ float sV[ND];
  __shared__ float sOut[32];
  const int tid = threadIdx.x, lane = tid & 31, wv = tid >> 5;
  const int b = blockIdx.x, t0 = blockIdx.y * 32;
  const int wmRow = (wv >> 2) * 16;
  const int wnCol = (wv & 3) * 64;

  // annots (32 x 512) fp32 -> bf16 LDS: 8 floats per step -> 1 uint4 LDS store.
  const float* Ag = annots + ((size_t)b * NT + t0) * NA;
  for (int idx = tid; idx < 32 * NA / 8; idx += 256) {
    int mm = idx >> 6, kq = (idx & 63) * 8;
    const float4* src = (const float4*)(Ag + (size_t)mm * NA + kq);
    float4 f0 = src[0], f1 = src[1];
    uint4 w;
    w.x = pk2(f0.x, f0.y);
    w.y = pk2(f0.z, f0.w);
    w.z = pk2(f1.x, f1.y);
    w.w = pk2(f1.z, f1.w);
    *(uint4*)&sA[mm * 520 + kq] = w;
  }
  // loc tile: contiguous bf16, re-strided to pad-40 rows (16B chunks).
  const unsigned short* Lg = loc_bf + ((size_t)b * NT + t0) * NF;
  if (tid < 128) {
    int mm = tid >> 2, q = tid & 3;
    uint4 w = *(const uint4*)(Lg + (size_t)mm * NF + q * 8);
    *(uint4*)&sLoc[mm * 40 + q * 8] = w;
  }
  if (tid < ND) {
    sPB[tid] = pq[(size_t)b * ND + tid] + a_b[tid] + loc_b[tid];
    sV[tid]  = v_w[tid];
  }
  if (tid < 32) sOut[tid] = 0.f;

  const v8f vz = {0.f, 0.f, 0.f, 0.f, 0.f, 0.f, 0.f, 0.f};
  v8f acc[4];
#pragma unroll
  for (int nt = 0; nt < 4; ++nt) acc[nt] = vz;

  constexpr int KSTEPS = NA / 32;  // 16 slabs for pa, +1 for ploc
  for (int s = 0; s <= KSTEPS; ++s) {
    const bool isLoc = (s == KSTEPS);
    __syncthreads();
    {  // contiguous b128 B-slab staging
      const uint4* srcB = (const uint4*)(isLoc ? locWp : (aWp + (size_t)s * 10240));
      uint4* dB = (uint4*)sB;
#pragma unroll
      for (int i = 0; i < 5; ++i) dB[tid + i * 256] = srcB[tid + i * 256];
      if (!isLoc && s + 1 < KSTEPS)
        __builtin_prefetch(aWp + (size_t)(s + 1) * 10240, 0, 1);
    }
    __syncthreads();

    BF16Frag fa;
    {
      const unsigned short* base = isLoc ? sLoc : sA;
      const int ld = isLoc ? 40 : 520;
      const int k0 = isLoc ? 0 : s * 32;
      const unsigned short* row =
          base + (wmRow + (lane & 15)) * ld + k0 + (lane >> 4) * 8;
#pragma unroll
      for (int g = 0; g < 2; ++g)
#pragma unroll
        for (int p = 0; p < 4; ++p)
          fa.u32[g * 4 + p] = *(const unsigned int*)(row + g * 16 + p * 2);
    }
#pragma unroll
    for (int nt = 0; nt < 4; ++nt) {
      BF16Frag fb;
      const unsigned short* row =
          sB + (wnCol + nt * 16 + (lane & 15)) * 40 + (lane >> 4) * 8;
#pragma unroll
      for (int p = 0; p < 8; ++p)
        fb.u32[p] = *(const unsigned int*)(row + (p >> 2) * 16 + (p & 3) * 2);
      acc[nt] = __builtin_amdgcn_wmma_f32_16x16x32_bf16(
          false, fa.v, false, fb.v, (short)0, acc[nt], false, false);
    }
  }
  __syncthreads();

  // tanh + v-weighted sum over d: 16-lane shfl reduction (one t-row per
  // accumulator register within a half-wave), then LDS atomics across the
  // 4 N-slice waves.
#pragma unroll
  for (int nt = 0; nt < 4; ++nt) {
    int n = wnCol + nt * 16 + (lane & 15);
    float pb = sPB[n], vv = sV[n];
#pragma unroll
    for (int r = 0; r < 8; ++r) {
      float val = tanhf(acc[nt][r] + pb) * vv;
      val += __shfl_xor(val, 1, 32);
      val += __shfl_xor(val, 2, 32);
      val += __shfl_xor(val, 4, 32);
      val += __shfl_xor(val, 8, 32);
      if ((lane & 15) == 0) {
        int m = wmRow + r + 8 * (lane >> 4);
        atomicAdd(&sOut[m], val);
      }
    }
  }
  __syncthreads();
  if (tid < 32) align_raw[(size_t)b * NT + t0 + tid] = sOut[tid];
}

// ---------------------------------------------------------------------------
// Masked softmax over T per batch. grid (B), 256 threads, 4 elems/thread.
// ---------------------------------------------------------------------------
__global__ __launch_bounds__(256) void k_softmax(
    const float* __restrict__ raw, const int* __restrict__ mask,
    float* __restrict__ out) {
  __shared__ float red[256];
  int b = blockIdx.x, tid = threadIdx.x;
  float v[4];
  float mx = -3.4e38f;
#pragma unroll
  for (int j = 0; j < 4; ++j) {
    int t = tid + j * 256;
    float x = raw[(size_t)b * NT + t];
    v[j] = (mask[(size_t)b * NT + t] != 0) ? x : -1e30f;
    mx = fmaxf(mx, v[j]);
  }
  red[tid] = mx;
  __syncthreads();
  for (int s = 128; s > 0; s >>= 1) {
    if (tid < s) red[tid] = fmaxf(red[tid], red[tid + s]);
    __syncthreads();
  }
  mx = red[0];
  __syncthreads();
  float sum = 0.f;
#pragma unroll
  for (int j = 0; j < 4; ++j) {
    v[j] = __expf(v[j] - mx);
    sum += v[j];
  }
  red[tid] = sum;
  __syncthreads();
  for (int s = 128; s > 0; s >>= 1) {
    if (tid < s) red[tid] += red[tid + s];
    __syncthreads();
  }
  float inv = 1.f / red[0];
#pragma unroll
  for (int j = 0; j < 4; ++j)
    out[(size_t)b * NT + tid + j * 256] = v[j] * inv;
}

// ---------------------------------------------------------------------------
// ctx[b,a] = sum_t align[b,t] * annots[b,t,a]. grid (B, A/256), coalesced in a.
// ---------------------------------------------------------------------------
__global__ __launch_bounds__(256) void k_ctx(
    const float* __restrict__ align, const float* __restrict__ annots,
    float* __restrict__ ctx) {
  int b = blockIdx.x;
  int a = blockIdx.y * 256 + threadIdx.x;
  const float* ab = align + (size_t)b * NT;
  const float* an = annots + (size_t)b * NT * NA + a;
  float acc = 0.f;
  for (int t = 0; t < NT; ++t) acc = fmaf(ab[t], an[(size_t)t * NA], acc);
  ctx[(size_t)b * NA + a] = acc;
}

// ---------------------------------------------------------------------------
extern "C" void kernel_launch(void* const* d_in, const int* in_sizes, int n_in,
                              void* d_out, int out_size, void* d_ws, size_t ws_size,
                              hipStream_t stream) {
  const float* memory  = (const float*)d_in[0];
  const float* context = (const float*)d_in[1];
  const float* rnn     = (const float*)d_in[2];
  const float* annots  = (const float*)d_in[3];
  const float* atten   = (const float*)d_in[4];
  const int*   mask    = (const int*)d_in[5];
  const float* Wi      = (const float*)d_in[6];
  const float* Wh      = (const float*)d_in[7];
  const float* bi      = (const float*)d_in[8];
  const float* bh      = (const float*)d_in[9];
  const float* conv_w  = (const float*)d_in[10];
  const float* loc_w   = (const float*)d_in[11];
  const float* loc_b   = (const float*)d_in[12];
  const float* q_w     = (const float*)d_in[13];
  const float* q_b     = (const float*)d_in[14];
  const float* a_w     = (const float*)d_in[15];
  const float* a_b     = (const float*)d_in[16];
  const float* v_w     = (const float*)d_in[17];

  float* out_h     = (float*)d_out;         // (B, R)
  float* out_ctx   = out_h + NB * NR;       // (B, A)
  float* out_align = out_ctx + NB * NA;     // (B, T)

  // Workspace carving (~27 MB total).
  char*  ws  = (char*)d_ws;
  size_t off = 0;
  auto carve = [&](size_t bytes) -> void* {
    void* p = ws + off;
    off += (bytes + 255) & ~(size_t)255;
    return p;
  };
  // Packed B weights: NBLK*(K/32) slabs of 10240 ushorts.
  unsigned short* WiTp   = (unsigned short*)carve((size_t)12 * 24 * 10240 * 2);
  unsigned short* WhTp   = (unsigned short*)carve((size_t)12 * 32 * 10240 * 2);
  unsigned short* aWp    = (unsigned short*)carve((size_t)1 * 16 * 10240 * 2);
  unsigned short* locWp  = (unsigned short*)carve((size_t)1 * 1 * 10240 * 2);
  unsigned short* qWp    = (unsigned short*)carve((size_t)1 * 32 * 10240 * 2);
  // Packed A activations: (M/32)*(K/32) slabs of 1280 ushorts.
  unsigned short* x_pk   = (unsigned short*)carve((size_t)4 * 24 * 1280 * 2);
  unsigned short* h0_pk  = (unsigned short*)carve((size_t)4 * 32 * 1280 * 2);
  unsigned short* h_pk   = (unsigned short*)carve((size_t)4 * 32 * 1280 * 2);
  float*          gi     = (float*)carve((size_t)NB * 3 * NR * 4);
  float*          gh     = (float*)carve((size_t)NB * 3 * NR * 4);
  float*          pq     = (float*)carve((size_t)NB * ND * 4);
  unsigned short* loc_bf = (unsigned short*)carve((size_t)NB * NT * NF * 2);
  float*          a_raw  = (float*)carve((size_t)NB * NT * 4);
  (void)ws_size; (void)in_sizes; (void)n_in; (void)out_size;

  // 1) Pack weights into fragment-slab bf16 layout (one-time transpose cost).
  k_packB<<<dim3(768 / 32, 3072 / 32), 256, 0, stream>>>(Wi, WiTp, 3072, 768);
  k_packB<<<dim3(1024 / 32, 3072 / 32), 256, 0, stream>>>(Wh, WhTp, 3072, 1024);
  k_packB<<<dim3(NA / 32, ND / 32), 256, 0, stream>>>(a_w, aWp, ND, NA);
  k_packB<<<dim3(NF / 32, ND / 32), 256, 0, stream>>>(loc_w, locWp, ND, NF);
  k_packB<<<dim3(NR / 32, ND / 32), 256, 0, stream>>>(q_w, qWp, ND, NR);

  // 2) Activation prep (packed-A).
  k_prep<<<(NB * NR + 255) / 256, 256, 0, stream>>>(memory, context, rnn, x_pk, h0_pk);

  // 3) GRU gates via WMMA GEMMs, then elementwise GRU.
  k_gemm_bf16<<<dim3(NB / 32, 3072 / 256), 256, 0, stream>>>(x_pk, WiTp, bi, gi, NB, 3072, 768);
  k_gemm_bf16<<<dim3(NB / 32, 3072 / 256), 256, 0, stream>>>(h0_pk, WhTp, bh, gh, NB, 3072, 1024);
  k_gru<<<(NB * NR + 255) / 256, 256, 0, stream>>>(gi, gh, rnn, out_h, h_pk);

  // 4) pq = h @ q_w.T + q_b (WMMA), and location conv (independent).
  k_gemm_bf16<<<dim3(NB / 32, ND / 256), 256, 0, stream>>>(h_pk, qWp, q_b, pq, NB, ND, NR);
  k_locconv<<<dim3(NB, NT / 128), 256, 0, stream>>>(atten, conv_w, loc_bf);

  // 5) Fused alignment logits (dominant WMMA kernel).
  k_align<<<dim3(NB, NT / 32), 256, 0, stream>>>(annots, aWp, locWp, loc_bf, pq,
                                                 a_b, loc_b, v_w, a_raw);

  // 6) Masked softmax, 7) context.
  k_softmax<<<NB, 256, 0, stream>>>(a_raw, mask, out_align);
  k_ctx<<<dim3(NB, NA / 256), 256, 0, stream>>>(out_align, annots, out_ctx);
}
